// DenseGraphConvEdgeToEdge_41583873360240
// MI455X (gfx1250) — matrix operand order repaired and checked by reading, not compiled, not run on hardware
//
#include <hip/hip_runtime.h>

// Problem constants (match reference)
#define Bt 4
#define Nn 256
#define Cc 128
#define Oo 128
#define CO (Cc * Oo)   // 16384

typedef __attribute__((ext_vector_type(16))) _Float16 v16h;
typedef __attribute__((ext_vector_type(8)))  _Float16 v8h;
typedef __attribute__((ext_vector_type(8)))  float    v8f;

// ---------------------------------------------------------------------------
// Kernel 1a: sr[b,i,c] = sum_j E[b,i,j,c]      (one block per (b,i), c = tid)
// ---------------------------------------------------------------------------
__global__ __launch_bounds__(Cc) void k_sumrows(const float* __restrict__ E,
                                                float* __restrict__ sr) {
  const int bn = blockIdx.x;            // b*N + i
  const int c  = threadIdx.x;
  const float* base = E + (size_t)bn * Nn * Cc + c;
  float acc = 0.f;
  #pragma unroll 4
  for (int j = 0; j < Nn; ++j) acc += base[(size_t)j * Cc];
  sr[(size_t)bn * Cc + c] = acc;
}

// ---------------------------------------------------------------------------
// Kernel 1b: sc[b,j,c] = sum_i E[b,i,j,c]      (one block per (b,j), c = tid)
// E resides in L2 (128 MiB < 192 MB) after k_sumrows -> cheap second pass.
// ---------------------------------------------------------------------------
__global__ __launch_bounds__(Cc) void k_sumcols(const float* __restrict__ E,
                                                float* __restrict__ sc) {
  const int b = blockIdx.x / Nn;
  const int j = blockIdx.x % Nn;
  const int c = threadIdx.x;
  const float* base = E + (size_t)b * Nn * Nn * Cc + (size_t)j * Cc + c;
  float acc = 0.f;
  #pragma unroll 4
  for (int i = 0; i < Nn; ++i) acc += base[(size_t)i * Nn * Cc];
  sc[(size_t)(b * Nn + j) * Cc + c] = acc;
}

// ---------------------------------------------------------------------------
// Kernel 2a: Ri[b,n,o] = sr@W2 + sc@W5 ; Cj[b,n,o] = sc@W3 + sr@W4
// one block per (b,n), o = tid. Tiny GEMMs (0.13 GFLOP total), VALU is fine.
// ---------------------------------------------------------------------------
__global__ __launch_bounds__(Oo) void k_ricj(const float* __restrict__ sr,
                                             const float* __restrict__ sc,
                                             const float* __restrict__ W,
                                             float* __restrict__ Ri,
                                             float* __restrict__ Cj) {
  const int bn = blockIdx.x;
  const int o  = threadIdx.x;
  const float* srp = sr + (size_t)bn * Cc;
  const float* scp = sc + (size_t)bn * Cc;
  const float* W2 = W + 2 * CO;
  const float* W3 = W + 3 * CO;
  const float* W4 = W + 4 * CO;
  const float* W5 = W + 5 * CO;
  float ri = 0.f, cj = 0.f;
  #pragma unroll 4
  for (int c = 0; c < Cc; ++c) {
    const float s_r = srp[c];            // scalar broadcast loads
    const float s_c = scp[c];
    ri += s_r * W2[c * Oo + o] + s_c * W5[c * Oo + o];
    cj += s_c * W3[c * Oo + o] + s_r * W4[c * Oo + o];
  }
  Ri[(size_t)bn * Oo + o] = ri;
  Cj[(size_t)bn * Oo + o] = cj;
}

// ---------------------------------------------------------------------------
// Kernel 2b: Kb[b,o] = (sum_n sr[b,n,:]) @ W6 + bias      (one block per b)
// ---------------------------------------------------------------------------
__global__ __launch_bounds__(Oo) void k_kb(const float* __restrict__ sr,
                                           const float* __restrict__ W,
                                           const float* __restrict__ bias,
                                           float* __restrict__ Kb) {
  __shared__ float sa[Cc];
  const int b = blockIdx.x;
  const int t = threadIdx.x;             // serves as both c and o (C==O)
  float s = 0.f;
  for (int n = 0; n < Nn; ++n) s += sr[(size_t)(b * Nn + n) * Cc + t];
  sa[t] = s;
  __syncthreads();
  const float* W6 = W + 6 * CO;
  float acc = bias[t];
  #pragma unroll 4
  for (int c = 0; c < Cc; ++c) acc += sa[c] * W6[c * Oo + t];
  Kb[(size_t)b * Oo + t] = acc;
}

// ---------------------------------------------------------------------------
// Kernel 2c: f16 copy of W0|W1 (row-major [c][o]) for the WMMA B operands
// ---------------------------------------------------------------------------
__global__ __launch_bounds__(256) void k_wh(const float* __restrict__ W,
                                            _Float16* __restrict__ Wh) {
  const int idx = blockIdx.x * 256 + threadIdx.x;   // 0 .. 2*CO-1
  Wh[idx] = (_Float16)W[idx];
}

// ---------------------------------------------------------------------------
// Main kernel helpers
// ---------------------------------------------------------------------------

// A-matrix 16x32 f16 lane layout (ISA 7.12.2):
//   lane<16 : elements 0..7 = K kb+0..7,   8..15 = K kb+16..23
//   lane>=16: elements 0..7 = K kb+8..15,  8..15 = K kb+24..31
// r points at this lane's M-row (f32, contiguous in c).
__device__ inline v16h cvt_rowA(const float* __restrict__ r, int kb, int sel) {
  const float4* p0 = (const float4*)(r + kb + sel * 8);        // 32B aligned
  const float4* p1 = (const float4*)(r + kb + 16 + sel * 8);
  const float4 a = p0[0], bq = p0[1], cq = p1[0], dq = p1[1];
  v16h A;
  A[0]  = (_Float16)a.x;  A[1]  = (_Float16)a.y;
  A[2]  = (_Float16)a.z;  A[3]  = (_Float16)a.w;
  A[4]  = (_Float16)bq.x; A[5]  = (_Float16)bq.y;
  A[6]  = (_Float16)bq.z; A[7]  = (_Float16)bq.w;
  A[8]  = (_Float16)cq.x; A[9]  = (_Float16)cq.y;
  A[10] = (_Float16)cq.z; A[11] = (_Float16)cq.w;
  A[12] = (_Float16)dq.x; A[13] = (_Float16)dq.y;
  A[14] = (_Float16)dq.z; A[15] = (_Float16)dq.w;
  return A;
}

// B-matrix 32x16 f16 layout: lane l holds column N = obase + (l & 15);
// elements 0..7 = K kb+8h+0..7, 8..15 = K kb+16+8h+0..7 (h = l >> 4).
__device__ inline v16h loadB(const _Float16* __restrict__ Wh, int kb,
                             int obase, int lane) {
  const _Float16* p = Wh + (size_t)(kb + 8 * (lane >> 4)) * Oo + obase + (lane & 15);
  v16h Bv;
  #pragma unroll
  for (int k = 0; k < 8; ++k) { Bv[k] = p[(size_t)k * Oo]; Bv[8 + k] = p[(size_t)(16 + k) * Oo]; }
  return Bv;
}

// ---------------------------------------------------------------------------
// Kernel 3 (main): one wave per 16-row M-tile (rows = j_base..j_base+15 for a
// fixed (b,i)), full O=128. 64 x v_wmma_f32_16x16x32_f16 per wave.
// ---------------------------------------------------------------------------
__global__ __launch_bounds__(256) void k_main(const float* __restrict__ E,
                                              const _Float16* __restrict__ Wh,
                                              const float* __restrict__ Ri,
                                              const float* __restrict__ Cj,
                                              const float* __restrict__ Kb,
                                              float* __restrict__ out) {
  const int wave = threadIdx.x >> 5;
  const int lane = threadIdx.x & 31;
  const int t = blockIdx.x * 8 + wave;        // tile id, 0..16383
  const int jt = t & 15;
  const int i  = (t >> 4) & (Nn - 1);
  const int b  = t >> 12;
  const int j_base = jt << 4;
  const int m   = lane & 15;
  const int sel = lane >> 4;

  // Per-lane A-row pointers: term0 rows E[b,i,j_base+m,:] (contiguous),
  // term1 rows E[b,j_base+m,i,:] (strided, L2-resident).
  const float* r0 = E + (((size_t)b * Nn + i) * Nn + (j_base + m)) * Cc;
  const float* r1 = E + (((size_t)b * Nn + (j_base + m)) * Nn + i) * Cc;
  const _Float16* Wh1 = Wh + CO;

  v8f acc[8];
  #pragma unroll
  for (int ob = 0; ob < 8; ++ob) acc[ob] = (v8f)(0.f);

  #pragma unroll
  for (int kc = 0; kc < 4; ++kc) {
    const int kb = kc * 32;
    const v16h A0 = cvt_rowA(r0, kb, sel);
    const v16h A1 = cvt_rowA(r1, kb, sel);
    #pragma unroll
    for (int ob = 0; ob < 8; ++ob) {
      const v16h B0 = loadB(Wh,  kb, ob * 16, lane);
      const v16h B1 = loadB(Wh1, kb, ob * 16, lane);
      acc[ob] = __builtin_amdgcn_wmma_f32_16x16x32_f16(
          false, A0, false, B0, (short)0, acc[ob], false, false);
      acc[ob] = __builtin_amdgcn_wmma_f32_16x16x32_f16(
          false, A1, false, B1, (short)0, acc[ob], false, false);
    }
  }

  // Epilogue: C/D layout -> VGPR v, lane<16: M=v N=lane; lane>=16: M=8+v.
  const int ncol = lane & 15;
  #pragma unroll
  for (int ob = 0; ob < 8; ++ob) {
    const int oc = ob * 16 + ncol;
    const float riv = Ri[((size_t)b * Nn + i) * Oo + oc] + Kb[(size_t)b * Oo + oc];
    #pragma unroll
    for (int v = 0; v < 8; ++v) {
      const int j = j_base + 8 * sel + v;
      const float val = acc[ob][v] + riv + Cj[((size_t)b * Nn + j) * Oo + oc];
      out[(((size_t)b * Nn + i) * Nn + j) * Oo + oc] = val;
    }
  }
}

// ---------------------------------------------------------------------------
// Launch
// ---------------------------------------------------------------------------
extern "C" void kernel_launch(void* const* d_in, const int* in_sizes, int n_in,
                              void* d_out, int out_size, void* d_ws, size_t ws_size,
                              hipStream_t stream) {
  // inputs: x (unused), adj (unused), edge_attrs, W, bias
  const float* E    = (const float*)d_in[2];
  const float* W    = (const float*)d_in[3];
  const float* bias = (const float*)d_in[4];
  float* out = (float*)d_out;

  // Workspace layout (floats), 16B-aligned f16 region at the end.
  float* wsf = (float*)d_ws;
  float* sr = wsf;                         // B*N*C = 131072
  float* sc = sr + (size_t)Bt * Nn * Cc;   // 131072
  float* Ri = sc + (size_t)Bt * Nn * Cc;   // B*N*O = 131072
  float* Cj = Ri + (size_t)Bt * Nn * Oo;   // 131072
  float* Kb = Cj + (size_t)Bt * Nn * Oo;   // B*O = 512
  _Float16* Wh = (_Float16*)(Kb + Bt * Oo); // 2*CO halves (offset 16B-aligned)

  k_sumrows<<<Bt * Nn, Cc, 0, stream>>>(E, sr);
  k_sumcols<<<Bt * Nn, Cc, 0, stream>>>(E, sc);
  k_ricj   <<<Bt * Nn, Oo, 0, stream>>>(sr, sc, W, Ri, Cj);
  k_kb     <<<Bt,      Oo, 0, stream>>>(sr, W, bias, Kb);
  k_wh     <<<(2 * CO) / 256, 256, 0, stream>>>(W, Wh);

  // 16384 M-tiles, 8 waves (one tile each) per 256-thread block
  k_main<<<(Bt * Nn * (Nn / 16)) / 8, 256, 0, stream>>>(E, Wh, Ri, Cj, Kb, out);
}
